// MambaBlock_67138928771575
// MI455X (gfx1250) — matrix operand order repaired and checked by reading, not compiled, not run on hardware
//
#include <hip/hip_runtime.h>
#include <hip/hip_bf16.h>
#include <math.h>

// ---------------------------------------------------------------------------
// Mamba block for MI455X (gfx1250): bf16 WMMA GEMMs + async-LDS fused scan.
// ---------------------------------------------------------------------------

typedef __attribute__((ext_vector_type(16))) __bf16 v16bf;
typedef __attribute__((ext_vector_type(8)))  __bf16 v8bf;
typedef __attribute__((ext_vector_type(8)))  float  v8f;

#define D_MODEL 1024
#define D_STATE 16
#define D_INNER 2048
#define B_SZ 4
#define L_SEQ 1024
#define M_ROWS (B_SZ * L_SEQ)          /* 4096 rows of every GEMM  */
#define N_XZ   (2 * D_INNER)           /* 4096                     */
#define N_DBC  (D_INNER + 2 * D_STATE) /* 2080                     */

// ---------------------------------------------------------------------------
// fp32 -> bf16 elementwise convert
// ---------------------------------------------------------------------------
__global__ void cvt_f32_bf16(const float* __restrict__ in,
                             __bf16* __restrict__ out, int n) {
  int i = blockIdx.x * blockDim.x + threadIdx.x;
  if (i < n) out[i] = (__bf16)in[i];
}

// ---------------------------------------------------------------------------
// A-fragment loader: 16x32 bf16 tile of row-major A (ld = K).
// Lane l: m = row0 + (l & 15), half = l >> 4.
//   elements 0..7  = A[m, k0 + half*8 + 0..7]
//   elements 8..15 = A[m, k0 + 16 + half*8 + 0..7]
// p must already point at A[m, k0 + half*8].
// ---------------------------------------------------------------------------
__device__ __forceinline__ v16bf load_a_frag(const __bf16* __restrict__ p) {
  v8bf lo = *(const v8bf*)p;
  v8bf hi = *(const v8bf*)(p + 16);
  return __builtin_shufflevector(lo, hi, 0, 1, 2, 3, 4, 5, 6, 7,
                                 8, 9, 10, 11, 12, 13, 14, 15);
}

__device__ __forceinline__ v8f wmma_bf16(v16bf a, v16bf b, v8f c) {
  return __builtin_amdgcn_wmma_f32_16x16x32_bf16(
      /*neg_a=*/false, a, /*neg_b=*/false, b,
      /*c_mod=*/(short)0, c, /*reuse_a=*/false, /*reuse_b=*/false);
}

// ---------------------------------------------------------------------------
// GEMM: C[M,N] (f32) = A[M,K] * W[N,K]^T, A/W bf16, fp32 accumulate.
// Block = 256 threads = 8 waves (4 in M x 2 in N) -> 128x128 block tile.
// Each wave: 32(M) x 64(N) strip, 8 accumulators, straight-line K loop
// (2 A-frag loads + 4 B-frag loads -> 8 v_wmma per K-step of 32).
// N handling: N is always a multiple of 16; waves whose 64-wide strip would
// overrun are clamped to the last in-range strip and recompute/rewrite
// identical values (deterministic duplicate stores, no in-loop branches).
// MODE 0: plain store.  MODE 1: C = softplus(acc + bias[n]).
// ---------------------------------------------------------------------------
template <int MODE>
__global__ __launch_bounds__(256) void gemm_bf16_wmma(
    const __bf16* __restrict__ A, const __bf16* __restrict__ W,
    float* __restrict__ C, const float* __restrict__ bias,
    int M, int N, int K) {
  const int lane = threadIdx.x & 31;
  const int wave = threadIdx.x >> 5;
  const int wm = wave & 3;
  const int wn = wave >> 2;

  const int row0 = blockIdx.y * 128 + wm * 32;
  int col0 = blockIdx.x * 128 + wn * 64;
  if (col0 > N - 64) col0 = N - 64;  // uniform clamp (see note above)

  const int nl   = lane & 15;
  const int half = lane >> 4;

  // Per-lane base pointers; the 16*K tile strides fold into inst offsets.
  const __bf16* ap = A + (size_t)(row0 + nl) * K + half * 8;
  const __bf16* bp = W + (size_t)(col0 + nl) * K + half * 8;
  const size_t tstride = (size_t)16 * K;

  v8f acc[2][4] = {};

  for (int k = 0; k < K; k += 32) {
    v16bf a0 = load_a_frag(ap);
    v16bf a1 = load_a_frag(ap + tstride);
#pragma unroll
    for (int j = 0; j < 4; ++j) {
      v16bf b = load_a_frag(bp + (size_t)j * tstride);
      acc[0][j] = wmma_bf16(a0, b, acc[0][j]);
      acc[1][j] = wmma_bf16(a1, b, acc[1][j]);
    }
    ap += 32;
    bp += 32;
    __builtin_prefetch((const void*)(ap + 32), 0, 1);  // global_prefetch_b8
    __builtin_prefetch((const void*)(bp + 32), 0, 1);
  }

  // C/D layout: VGPR r, lane l -> row = r + 8*(l>>4), col = l&15
#pragma unroll
  for (int i = 0; i < 2; ++i) {
#pragma unroll
    for (int j = 0; j < 4; ++j) {
      const int n = col0 + j * 16 + nl;
#pragma unroll
      for (int r = 0; r < 8; ++r) {
        const int row = row0 + i * 16 + r + 8 * half;
        float v = acc[i][j][r];
        if (MODE == 1) {
          v += bias[n];
          v = (v > 20.f) ? v : log1pf(expf(v));  // softplus
        }
        C[(size_t)row * N + n] = v;
      }
    }
  }
}

// ---------------------------------------------------------------------------
// Causal depthwise conv (K=4) + bias + SiLU over xi = XZ[:, :, 0:D_INNER].
// One thread per (b, l, d). Emits f32 (for scan) and bf16 (for x_proj GEMM).
// ---------------------------------------------------------------------------
__global__ void conv_silu_kernel(const float* __restrict__ XZ,
                                 const float* __restrict__ convw,
                                 const float* __restrict__ convb,
                                 float* __restrict__ XI,
                                 __bf16* __restrict__ XIb) {
  const int idx = blockIdx.x * blockDim.x + threadIdx.x;  // B*L*D_INNER
  const int d  = idx & (D_INNER - 1);
  const int bl = idx >> 11;               // / D_INNER
  const int l  = bl & (L_SEQ - 1);
  const int b  = bl >> 10;

  float acc = convb[d];
#pragma unroll
  for (int k = 0; k < 4; ++k) {
    const int t = l + k - 3;              // causal: pad-left of 3
    if (t >= 0)
      acc += convw[d * 4 + k] * XZ[((size_t)(b * L_SEQ + t)) * N_XZ + d];
  }
  const float s = acc / (1.f + __expf(-acc));  // silu
  XI[idx]  = s;
  XIb[idx] = (__bf16)s;
}

// ---------------------------------------------------------------------------
// Extract delta_raw = DBC[:, 0:D_INNER] into a dense bf16 matrix for dt_proj.
// ---------------------------------------------------------------------------
__global__ void extract_delta_bf16(const float* __restrict__ DBC,
                                   __bf16* __restrict__ Db) {
  const int idx = blockIdx.x * blockDim.x + threadIdx.x;  // M_ROWS * D_INNER
  const int n = idx & (D_INNER - 1);
  const int m = idx >> 11;
  Db[idx] = (__bf16)DBC[(size_t)m * N_DBC + n];
}

// ---------------------------------------------------------------------------
// Selective scan, fused with output gating  y_g = y * silu(z)  -> bf16.
// One thread per (b, d) channel; 16-wide state in registers; Bp/Cp chunks
// staged into LDS with GLOBAL_LOAD_ASYNC_TO_LDS_B32 (ASYNCcnt-tracked,
// no VGPR round-trip), consumed by all 256 threads of the block.
// Grid: 32 blocks of 256  (b = blk>>3, d = (blk&7)*256 + tid).
// ---------------------------------------------------------------------------
#define L_CH 64
__global__ void scan_kernel(const float* __restrict__ DBC,
                            const float* __restrict__ DELTA,
                            const float* __restrict__ XI,
                            const float* __restrict__ XZ,
                            const float* __restrict__ A_log,
                            const float* __restrict__ Dp,
                            __bf16* __restrict__ Yb) {
  __shared__ float ldsBC[L_CH][2 * D_STATE];  // [l][0:16]=Bp, [16:32]=Cp

  const int tid = threadIdx.x;
  const int b = blockIdx.x >> 3;
  const int d = (blockIdx.x & 7) * 256 + tid;

  float An[D_STATE];
#pragma unroll
  for (int n = 0; n < D_STATE; ++n) An[n] = -__expf(A_log[d * D_STATE + n]);
  const float Dv = Dp[d];

  float h[D_STATE];
#pragma unroll
  for (int n = 0; n < D_STATE; ++n) h[n] = 0.f;

  for (int l0 = 0; l0 < L_SEQ; l0 += L_CH) {
    __syncthreads();  // previous chunk fully consumed before overwrite
    // Async global->LDS copy of this chunk's Bp/Cp (8 elements per thread).
    // Low 32 bits of a generic pointer into LDS are the wave-relative LDS
    // byte offset (ISA 10.2: LDS aperture maps addr[31:0] directly).
    for (int i = tid; i < L_CH * 2 * D_STATE; i += 256) {
      const int ll = i >> 5, j = i & 31;
      const float* gp =
          &DBC[((size_t)(b * L_SEQ + l0 + ll)) * N_DBC + D_INNER + j];
      const unsigned lds_off =
          (unsigned)(unsigned long long)&ldsBC[ll][j];
      asm volatile("global_load_async_to_lds_b32 %0, %1, off"
                   :: "v"(lds_off), "v"(gp)
                   : "memory");
    }
    asm volatile("s_wait_asynccnt 0x0" ::: "memory");
    __syncthreads();  // all waves' async copies landed in LDS

    for (int ll = 0; ll < L_CH; ++ll) {
      const size_t base = ((size_t)(b * L_SEQ + l0 + ll)) * D_INNER + d;
      const float delta = DELTA[base];
      const float xv    = XI[base];
      const float dx    = delta * xv;
      float y = 0.f;
#pragma unroll
      for (int n = 0; n < D_STATE; ++n) {
        const float dA = __expf(delta * An[n]);
        h[n] = dA * h[n] + dx * ldsBC[ll][n];
        y += ldsBC[ll][D_STATE + n] * h[n];
      }
      y += xv * Dv;
      const float z  = XZ[((size_t)(b * L_SEQ + l0 + ll)) * N_XZ + D_INNER + d];
      const float sz = z / (1.f + __expf(-z));
      Yb[base] = (__bf16)(y * sz);
    }
  }
}

// ---------------------------------------------------------------------------
// Host-side orchestration
// ---------------------------------------------------------------------------
extern "C" void kernel_launch(void* const* d_in, const int* in_sizes, int n_in,
                              void* d_out, int out_size, void* d_ws,
                              size_t ws_size, hipStream_t stream) {
  (void)in_sizes; (void)n_in; (void)out_size; (void)ws_size;

  const float* x          = (const float*)d_in[0];
  const float* in_proj_w  = (const float*)d_in[1];
  const float* conv_w     = (const float*)d_in[2];
  const float* conv_b     = (const float*)d_in[3];
  const float* x_proj_w   = (const float*)d_in[4];
  const float* dt_proj_w  = (const float*)d_in[5];
  const float* dt_proj_b  = (const float*)d_in[6];
  const float* A_log      = (const float*)d_in[7];
  const float* D_param    = (const float*)d_in[8];
  const float* out_proj_w = (const float*)d_in[9];
  float* out = (float*)d_out;

  // workspace layout (256B aligned); region0 reused Xb+W1b -> Db -> Yb
  char* w = (char*)d_ws;
  size_t off = 0;
  auto alloc = [&](size_t bytes) -> char* {
    char* p = w + off;
    off += (bytes + 255) & ~(size_t)255;
    return p;
  };
  char*   region0 = alloc((size_t)16 * 1024 * 1024);
  __bf16* Xb  = (__bf16*)region0;                         // 8 MB
  __bf16* W1b = (__bf16*)(region0 + 8388608);             // 8 MB
  __bf16* Db  = (__bf16*)region0;                         // 16 MB (reuse)
  __bf16* Yb  = (__bf16*)region0;                         // 16 MB (reuse)
  float*  XZ    = (float*) alloc((size_t)M_ROWS * N_XZ * 4);     // 64 MB
  float*  XI    = (float*) alloc((size_t)M_ROWS * D_INNER * 4);  // 32 MB
  __bf16* XIb   = (__bf16*)alloc((size_t)M_ROWS * D_INNER * 2);  // 16 MB
  __bf16* W2b   = (__bf16*)alloc((size_t)N_DBC * D_INNER * 2);   // 8.1 MB
  float*  DBC   = (float*) alloc((size_t)M_ROWS * N_DBC * 4);    // 32.5 MB
  __bf16* W3b   = (__bf16*)alloc((size_t)D_INNER * D_INNER * 2); // 8 MB
  float*  DELTA = (float*) alloc((size_t)M_ROWS * D_INNER * 4);  // 32 MB
  __bf16* W4b   = (__bf16*)alloc((size_t)D_MODEL * D_INNER * 2); // 4 MB

  const dim3 cvtB(256);
  auto cvt = [&](const float* src, __bf16* dst, int n) {
    cvt_f32_bf16<<<dim3((n + 255) / 256), cvtB, 0, stream>>>(src, dst, n);
  };

  // 1) bf16 conversions of activations + all four weight matrices
  cvt(x,          Xb,  M_ROWS * D_MODEL);
  cvt(in_proj_w,  W1b, N_XZ * D_MODEL);
  cvt(x_proj_w,   W2b, N_DBC * D_INNER);
  cvt(dt_proj_w,  W3b, D_INNER * D_INNER);
  cvt(out_proj_w, W4b, D_MODEL * D_INNER);

  const dim3 gblk(256);

  // 2) in_proj: XZ[4096,4096] = Xb * W1b^T
  gemm_bf16_wmma<0><<<dim3(N_XZ / 128, M_ROWS / 128), gblk, 0, stream>>>(
      Xb, W1b, XZ, nullptr, M_ROWS, N_XZ, D_MODEL);

  // 3) causal depthwise conv + SiLU
  conv_silu_kernel<<<dim3((M_ROWS * D_INNER) / 256), dim3(256), 0, stream>>>(
      XZ, conv_w, conv_b, XI, XIb);

  // 4) x_proj: DBC[4096,2080] = XIb * W2b^T
  gemm_bf16_wmma<0><<<dim3((N_DBC + 127) / 128, M_ROWS / 128), gblk, 0,
                     stream>>>(XIb, W2b, DBC, nullptr, M_ROWS, N_DBC, D_INNER);

  // 5) pack delta_raw -> bf16
  extract_delta_bf16<<<dim3((M_ROWS * D_INNER) / 256), dim3(256), 0, stream>>>(
      DBC, Db);

  // 6) dt_proj + bias + softplus: DELTA[4096,2048]
  gemm_bf16_wmma<1><<<dim3(D_INNER / 128, M_ROWS / 128), gblk, 0, stream>>>(
      Db, W3b, DELTA, dt_proj_b, M_ROWS, D_INNER, D_INNER);

  // 7) selective scan + gating -> Yb (bf16)
  scan_kernel<<<dim3((B_SZ * D_INNER) / 256), dim3(256), 0, stream>>>(
      DBC, DELTA, XI, XZ, A_log, D_param, Yb);

  // 8) out_proj: out[4096,1024] = Yb * W4b^T
  gemm_bf16_wmma<0><<<dim3(D_MODEL / 128, M_ROWS / 128), gblk, 0, stream>>>(
      Yb, W4b, out, nullptr, M_ROWS, D_MODEL, D_INNER);
}
